// Additive_attention_78219944395219
// MI455X (gfx1250) — matrix operand order repaired and checked by reading, not compiled, not run on hardware
//
#include <hip/hip_runtime.h>
#include <hip/hip_fp16.h>
#include <math.h>

typedef __attribute__((ext_vector_type(16))) _Float16 v16h;
typedef __attribute__((ext_vector_type(8)))  float    v8f;

#define B_  4
#define L_  1024
#define D_  256
#define U_  128
#define KS_ 16          // k-dimension split across blocks
#define KTILE 32        // K per WMMA f16 16x16x32
#define QSTR 132        // padded q-row stride in floats (avoid bank conflicts)

#if defined(__has_builtin)
#  if __has_builtin(__builtin_amdgcn_tanhf)
#    define TANH(x) __builtin_amdgcn_tanhf(x)
#  else
#    define TANH(x) tanhf(x)
#  endif
#else
#  define TANH(x) tanhf(x)
#endif

// ---------------- Kernel A: q = X@W1 + bias1, k = X@W2 ----------------
// grid: B*L blocks, 128 threads (one per u)
__global__ void proj_kernel(const float* __restrict__ X,
                            const float* __restrict__ W1,
                            const float* __restrict__ W2,
                            const float* __restrict__ bias1,
                            float* __restrict__ qs,
                            float* __restrict__ ks) {
    __shared__ float sX[D_];
    const int row = blockIdx.x;      // b*L + l
    const int u   = threadIdx.x;     // 0..127
    const float* xr = X + (size_t)row * D_;
    sX[u]      = xr[u];
    sX[u + U_] = xr[u + U_];
    __syncthreads();
    float a1 = 0.f, a2 = 0.f;
#pragma unroll 8
    for (int d = 0; d < D_; ++d) {
        const float x = sX[d];
        a1 = fmaf(x, W1[d * U_ + u], a1);
        a2 = fmaf(x, W2[d * U_ + u], a2);
    }
    qs[(size_t)row * U_ + u] = a1 + bias1[u];   // fold bias1 into q
    ks[(size_t)row * U_ + u] = a2;
}

// ---------------- Kernel B: zero output ----------------
__global__ void zero_kernel(float* __restrict__ p, int n) {
    const int i = blockIdx.x * blockDim.x + threadIdx.x;
    if (i < n) p[i] = 0.f;
}

// ---------------- Kernel C: scores + WMMA GEMM ----------------
// grid: (L/16, B, KS_), block: 32 threads (one wave)
__global__ void __launch_bounds__(32)
attn_kernel(const float* __restrict__ X,
            const float* __restrict__ qs,
            const float* __restrict__ ks,
            const float* __restrict__ W3,
            const float* __restrict__ bias2,
            float* __restrict__ out) {
    __shared__ float sQ[16 * QSTR];     // q tile, padded rows
    __shared__ float sK[KTILE * U_];    // k-projection tile
    __shared__ float sW3[U_];

    const int lane   = threadIdx.x;     // 0..31
    const int q0     = blockIdx.x * 16;
    const int b      = blockIdx.y;
    const int kz     = blockIdx.z;
    const int kspan  = L_ / KS_;        // 64 keys per block
    const int kstart = kz * kspan;

    // stage q tile (row-by-row into padded LDS): 16 rows x 32 float4
    {
        const float4* qsrc = (const float4*)(qs + ((size_t)(b * L_ + q0)) * U_);
#pragma unroll
        for (int i = 0; i < 16; ++i)
            ((float4*)(sQ + i * QSTR))[lane] = qsrc[i * 32 + lane];
        ((float4*)sW3)[lane] = ((const float4*)W3)[lane];
    }
    const float b2 = bias2[0];

    v8f acc[16];
#pragma unroll
    for (int n = 0; n < 16; ++n) acc[n] = (v8f){0.f,0.f,0.f,0.f,0.f,0.f,0.f,0.f};

    const int qrow  = lane & 15;
    const int khalf = lane >> 4;
    const float* qr = sQ + qrow * QSTR;

    for (int kt = 0; kt < kspan; kt += KTILE) {
        const int kbase = kstart + kt;
        __syncthreads();
        // stage k-projection tile: 32 rows x 128 floats = 1024 float4
        {
            const float4* ksrc = (const float4*)(ks + ((size_t)(b * L_ + kbase)) * U_);
            float4* sK4 = (float4*)sK;
#pragma unroll
            for (int i = 0; i < 32; ++i) sK4[lane + i * 32] = ksrc[lane + i * 32];
        }
        __syncthreads();

        // ---- additive scores: logits for this lane's 16 (q,k) pairs ----
        float sacc[16];
#pragma unroll
        for (int j = 0; j < 16; ++j) sacc[j] = 0.f;

        for (int u = 0; u < U_; u += 4) {
            const float4 q4 = *(const float4*)(qr + u);
            const float4 w4 = *(const float4*)(sW3 + u);
#pragma unroll
            for (int j = 0; j < 16; ++j) {
                // A-matrix 16-bit layout: lanes 0-15 hold K in {0..7,16..23},
                // lanes 16-31 hold K in {8..15,24..31}
                const int kloc = (j & 7) + khalf * 8 + ((j >> 3) << 4);
                const float4 k4 = *(const float4*)(sK + kloc * U_ + u);
                float s = sacc[j];
                s = fmaf(TANH(q4.x + k4.x), w4.x, s);
                s = fmaf(TANH(q4.y + k4.y), w4.y, s);
                s = fmaf(TANH(q4.z + k4.z), w4.z, s);
                s = fmaf(TANH(q4.w + k4.w), w4.w, s);
                sacc[j] = s;
            }
        }

        // sigmoid gate -> f16 A matrix (already in WMMA A layout order)
        v16h amat;
#pragma unroll
        for (int j = 0; j < 16; ++j) {
            const float x  = sacc[j] + b2;
            const float sg = 1.0f / (1.0f + __expf(-x));
            amat[j] = (_Float16)sg;
        }

        // ---- out[16q, 256d] += score(16x32) @ X(32x16 per N-chunk) ----
        // B layout (16-bit, K=32): lanes 0-15 hold K in {0..7,16..23}, lanes 16-31 K in {8..15,24..31},
        // column N = lane & 15
        const float* Xk = X + ((size_t)(b * L_ + kbase + khalf * 8)) * D_ + (lane & 15);
#pragma unroll
        for (int n = 0; n < 16; ++n) {
            v16h bmat;
#pragma unroll
            for (int m = 0; m < 16; ++m)
                bmat[m] = (_Float16)Xk[(size_t)((m & 7) + ((m >> 3) << 4)) * D_ + n * 16];
            acc[n] = __builtin_amdgcn_wmma_f32_16x16x32_f16(
                false, amat, false, bmat, (short)0, acc[n], false, false);
        }
    }

    // ---- accumulate partial output (k-split) via f32 atomics ----
    // C/D layout: VGPR r, lanes 0-15 -> M=r, lanes 16-31 -> M=r+8; N = lane&15
    const int col = lane & 15;
#pragma unroll
    for (int n = 0; n < 16; ++n) {
#pragma unroll
        for (int r = 0; r < 8; ++r) {
            const int M = r + khalf * 8;
            atomicAdd(&out[((size_t)(b * L_ + q0 + M)) * D_ + n * 16 + col], acc[n][r]);
        }
    }
}

// ---------------- launcher ----------------
extern "C" void kernel_launch(void* const* d_in, const int* in_sizes, int n_in,
                              void* d_out, int out_size, void* d_ws, size_t ws_size,
                              hipStream_t stream) {
    const float* X     = (const float*)d_in[0];
    const float* W1    = (const float*)d_in[1];
    const float* W2    = (const float*)d_in[2];
    const float* W3    = (const float*)d_in[3];
    const float* bias1 = (const float*)d_in[4];
    const float* bias2 = (const float*)d_in[5];
    float* out = (float*)d_out;

    float* qs = (float*)d_ws;                       // [B,L,U] f32 (bias folded)
    float* kk = qs + (size_t)B_ * L_ * U_;          // [B,L,U] f32

    proj_kernel<<<B_ * L_, U_, 0, stream>>>(X, W1, W2, bias1, qs, kk);

    const int n = B_ * L_ * D_;
    zero_kernel<<<(n + 255) / 256, 256, 0, stream>>>(out, n);

    attn_kernel<<<dim3(L_ / 16, B_, KS_), 32, 0, stream>>>(X, qs, kk, W3, bias2, out);
}
